// ConcatSquashGNN_42752104464516
// MI455X (gfx1250) — compile-verified, matrix operand order checked by reading, not run on hardware
//
#include <hip/hip_runtime.h>
#include <hip/hip_bf16.h>
#include <math.h>

typedef __attribute__((ext_vector_type(16))) __bf16 v16bf;
typedef __attribute__((ext_vector_type(8)))  float  v8f;

// ---------------------------------------------------------------------------
// hyper-network: gate = sigmoid(ctx @ Wg + bg), hbias = ctx @ Wb
// ctx: [1,64], Wg/Wb: [64,128] row-major
// ---------------------------------------------------------------------------
__global__ __launch_bounds__(128) void hyper_kernel(
    const float* __restrict__ ctx, const float* __restrict__ Wg,
    const float* __restrict__ bg,  const float* __restrict__ Wb,
    float* __restrict__ gate, float* __restrict__ hbias) {
  int j = threadIdx.x;                 // output channel 0..127
  float g = bg[j], bb = 0.0f;
#pragma unroll 8
  for (int k = 0; k < 64; ++k) {
    float c = ctx[k];
    g  += c * Wg[k * 128 + j];
    bb += c * Wb[k * 128 + j];
  }
  gate[j]  = 1.0f / (1.0f + __expf(-g));
  hbias[j] = bb;
}

// ---------------------------------------------------------------------------
// degree / norm kernels: deg[i] = 1 (self loop) + #incoming edges; dinv=rsqrt
// ---------------------------------------------------------------------------
__global__ void deg_init_kernel(float* __restrict__ deg, int n) {
  int i = blockIdx.x * blockDim.x + threadIdx.x;
  if (i < n) deg[i] = 1.0f;
}

__global__ void deg_accum_kernel(const int* __restrict__ dst,
                                 float* __restrict__ deg, int e) {
  int i = blockIdx.x * blockDim.x + threadIdx.x;
  if (i < e)
    __hip_atomic_fetch_add(&deg[dst[i]], 1.0f, __ATOMIC_RELAXED,
                           __HIP_MEMORY_SCOPE_AGENT);
}

__global__ void deg_finalize_kernel(float* __restrict__ deg, int n) {
  int i = blockIdx.x * blockDim.x + threadIdx.x;
  if (i < n) deg[i] = rsqrtf(deg[i]);
}

// ---------------------------------------------------------------------------
// GEMM: H[nrows,128] = act(A[nrows,128] (+bias)) @ W[128,128]
// bf16 WMMA, f32 accumulate. One 128-row tile per 256-thread block.
// applyAct=1 fuses  v = leaky_relu(v + bias[k], 0.2)  into the tile load.
// ---------------------------------------------------------------------------
__global__ __launch_bounds__(256) void gemm_wmma_kernel(
    const float* __restrict__ A, const float* __restrict__ W,
    const float* __restrict__ bias, float* __restrict__ H,
    int nrows, int applyAct) {
  __shared__ __bf16 sA[128 * 128];   // A tile, row-major [row][k]      (32 KB)
  __shared__ __bf16 sW[128 * 128];   // W transposed,   [n][k]          (32 KB)

  const int tid = threadIdx.x;
  const int rowbase = blockIdx.x << 7;

  // stage W^T into LDS as bf16
  for (int i = tid; i < 128 * 128; i += 256) {
    int k = i >> 7, n = i & 127;
    sW[n * 128 + k] = (__bf16)W[i];
  }
  // stage A tile (zero-padded past nrows), fused bias + leaky relu for layer 2
  for (int i = tid; i < 128 * 128; i += 256) {
    int r = i >> 7, k = i & 127;
    int row = rowbase + r;
    float v = 0.0f;
    if (row < nrows) {
      v = A[(size_t)row * 128 + k];
      if (applyAct) {
        v += bias[k];
        v = fmaxf(v, 0.2f * v);      // leaky_relu, slope 0.2
      }
    }
    sA[i] = (__bf16)v;
  }
  __syncthreads();

  const int w       = tid >> 5;        // wave 0..7  -> rows w*16..w*16+15
  const int lane    = tid & 31;
  const int lrow    = lane & 15;
  const int halfsel = lane >> 4;       // 0: K-base 0, 1: K-base +8
  const int m       = (w << 4) + lrow; // A row within tile for this lane

  v8f acc[8] = {};                     // 8 column tiles of 16

#pragma unroll
  for (int kc = 0; kc < 4; ++kc) {     // K = 128 in 4 chunks of 32
    const int kb = kc * 32 + halfsel * 8;

    // A fragment: elems 0..7 = K kb..kb+7, elems 8..15 = K kb+16..kb+23
    v16bf afrag;
    const __bf16* ap = &sA[m * 128 + kb];
#pragma unroll
    for (int e = 0; e < 8; ++e) { afrag[e] = ap[e]; afrag[e + 8] = ap[e + 16]; }

#pragma unroll
    for (int t = 0; t < 8; ++t) {
      // B fragment: lane owns column n = t*16 + lrow, contiguous K in sW[n][*]
      v16bf bfrag;
      const __bf16* bp = &sW[(t * 16 + lrow) * 128 + kb];
#pragma unroll
      for (int e = 0; e < 8; ++e) { bfrag[e] = bp[e]; bfrag[e + 8] = bp[e + 16]; }

      acc[t] = __builtin_amdgcn_wmma_f32_16x16x32_bf16(
          false, afrag, false, bfrag, (short)0, acc[t], false, false);
    }
  }

  // C/D layout: VGPR r -> M = r (lanes 0-15) or r+8 (lanes 16-31), N = lane&15
  const int rbase = rowbase + (w << 4) + halfsel * 8;
#pragma unroll
  for (int t = 0; t < 8; ++t) {
    int n = t * 16 + lrow;
#pragma unroll
    for (int r = 0; r < 8; ++r) {
      int row = rbase + r;
      if (row < nrows) H[(size_t)row * 128 + n] = acc[t][r];
    }
  }
}

// ---------------------------------------------------------------------------
// acc[i] = dinv[row]^2 * h[i]   (self-loop contribution; also initializes acc)
// ---------------------------------------------------------------------------
__global__ void self_init_kernel(const float* __restrict__ h,
                                 const float* __restrict__ dinv,
                                 float* __restrict__ acc, int total) {
  int i = blockIdx.x * blockDim.x + threadIdx.x;
  if (i < total) {
    int row = i >> 7;
    float di = dinv[row];
    acc[i] = di * di * h[i];
  }
}

// ---------------------------------------------------------------------------
// scatter: one wave32 per edge, each lane handles a float4 of channels.
// acc[dst] += dinv[src]*dinv[dst] * h[src]
// ---------------------------------------------------------------------------
__global__ __launch_bounds__(256) void scatter_kernel(
    const float* __restrict__ h, const int* __restrict__ src,
    const int* __restrict__ dst, const float* __restrict__ dinv,
    float* __restrict__ acc, int e) {
  int edge = blockIdx.x * 8 + (threadIdx.x >> 5);
  if (edge >= e) return;
  int lane = threadIdx.x & 31;
  int s = src[edge], d = dst[edge];
  float nrm = dinv[s] * dinv[d];
  const float4 hv = *(const float4*)(&h[(size_t)s * 128 + lane * 4]);
  float* out = &acc[(size_t)d * 128 + lane * 4];
  __hip_atomic_fetch_add(out + 0, hv.x * nrm, __ATOMIC_RELAXED, __HIP_MEMORY_SCOPE_AGENT);
  __hip_atomic_fetch_add(out + 1, hv.y * nrm, __ATOMIC_RELAXED, __HIP_MEMORY_SCOPE_AGENT);
  __hip_atomic_fetch_add(out + 2, hv.z * nrm, __ATOMIC_RELAXED, __HIP_MEMORY_SCOPE_AGENT);
  __hip_atomic_fetch_add(out + 3, hv.w * nrm, __ATOMIC_RELAXED, __HIP_MEMORY_SCOPE_AGENT);
}

// ---------------------------------------------------------------------------
// out[i] = (out[i] + b2[c]) * gate[c] + hbias[c]
// ---------------------------------------------------------------------------
__global__ void finalize_kernel(float* __restrict__ out,
                                const float* __restrict__ b2,
                                const float* __restrict__ gate,
                                const float* __restrict__ hbias, int total) {
  int i = blockIdx.x * blockDim.x + threadIdx.x;
  if (i < total) {
    int c = i & 127;
    out[i] = (out[i] + b2[c]) * gate[c] + hbias[c];
  }
}

// ---------------------------------------------------------------------------
extern "C" void kernel_launch(void* const* d_in, const int* in_sizes, int n_in,
                              void* d_out, int out_size, void* d_ws, size_t ws_size,
                              hipStream_t stream) {
  const float* x   = (const float*)d_in[0];
  const float* ctx = (const float*)d_in[1];
  const float* W1  = (const float*)d_in[2];
  const float* b1  = (const float*)d_in[3];
  const float* W2  = (const float*)d_in[4];
  const float* b2  = (const float*)d_in[5];
  const float* Wg  = (const float*)d_in[6];
  const float* bg  = (const float*)d_in[7];
  const float* Wb  = (const float*)d_in[8];
  const int*   ei  = (const int*)d_in[9];

  const int N = in_sizes[0] / 128;
  const int E = in_sizes[9] / 2;
  const int* srcp = ei;
  const int* dstp = ei + E;

  float* ws    = (float*)d_ws;
  float* gate  = ws;                      // 128
  float* hbias = ws + 128;                // 128
  float* dinv  = ws + 256;                // N (padded to 50048)
  float* h1    = ws + 256 + 50048;        // N*128
  float* acc1  = h1 + (size_t)N * 128;    // N*128
  float* h2    = acc1 + (size_t)N * 128;  // N*128
  float* outp  = (float*)d_out;

  const int total  = N * 128;
  const int gtiles = (N + 127) / 128;

  // hyper gate/bias
  hyper_kernel<<<1, 128, 0, stream>>>(ctx, Wg, bg, Wb, gate, hbias);

  // symmetric normalization: deg (with self loop) -> dinv
  deg_init_kernel<<<(N + 255) / 256, 256, 0, stream>>>(dinv, N);
  deg_accum_kernel<<<(E + 255) / 256, 256, 0, stream>>>(dstp, dinv, E);
  deg_finalize_kernel<<<(N + 255) / 256, 256, 0, stream>>>(dinv, N);

  // layer 1: h1 = x @ W1  (WMMA), acc1 = scatter(norm * h1) incl self loop
  gemm_wmma_kernel<<<gtiles, 256, 0, stream>>>(x, W1, nullptr, h1, N, 0);
  self_init_kernel<<<(total + 255) / 256, 256, 0, stream>>>(h1, dinv, acc1, total);
  scatter_kernel<<<(E + 7) / 8, 256, 0, stream>>>(h1, srcp, dstp, dinv, acc1, E);

  // layer 2: h2 = leaky_relu(acc1 + b1) @ W2  (bias+act fused into tile load)
  gemm_wmma_kernel<<<gtiles, 256, 0, stream>>>(acc1, W2, b1, h2, N, 1);
  self_init_kernel<<<(total + 255) / 256, 256, 0, stream>>>(h2, dinv, outp, total);
  scatter_kernel<<<(E + 7) / 8, 256, 0, stream>>>(h2, srcp, dstp, dinv, outp, E);

  // out = (acc2 + b2) * gate + hbias
  finalize_kernel<<<(total + 255) / 256, 256, 0, stream>>>(outp, b2, gate, hbias, total);
}
